// ExtendedMemory_69441031242178
// MI455X (gfx1250) — matrix operand order repaired and checked by reading, not compiled, not run on hardware
//
#include <hip/hip_runtime.h>
#include <hip/hip_fp16.h>
#include <cstdint>

typedef __attribute__((ext_vector_type(16))) _Float16 v16h;
typedef __attribute__((ext_vector_type(8)))  _Float16 v8h;
typedef __attribute__((ext_vector_type(8)))  float    v8f;

// Model constants (fixed by the reference)
#define Lc   2
#define Dc   1024
#define HSc  64
#define Hc   16
#define Ec   32
#define EDc  64
#define Bc   4
#define Tc   2048
#define BTc  (Bc * Tc)
#define EPSc 1e-5f
#define CTc  128              // wkv chunk length
#define NCc  (Tc / CTc)       // 16 chunks

// ---------------------------------------------------------------------------
// WMMA GEMM: C[M,N] = A[M,K] (f16, row-major, lda) x B (f16, stored transposed
// as Bt[N][K], row stride ldb) with f32 accumulation and a fused epilogue EP.
// Block: 256 threads = 8 waves (4x2), block tile 128x128, wave tile 32x64.
// ---------------------------------------------------------------------------
template<int EP>
__global__ __launch_bounds__(256)
void gemm_wmma_f16(const _Float16* __restrict__ A, int lda,
                   const _Float16* __restrict__ Bt, int ldb,
                   float* __restrict__ Cf, _Float16* __restrict__ Ch, int ldc,
                   int M, int N, int K,
                   const float* __restrict__ e0,
                   const float* __restrict__ e1,
                   const float* __restrict__ e2)
{
    const int wave = threadIdx.x >> 5;
    const int lane = threadIdx.x & 31;
    const int ll   = lane & 15;       // low-lane index (M or N within 16)
    const int hi   = lane >> 4;       // lane group 0/1
    const int wm = wave & 3, wn = wave >> 2;
    const int m0 = blockIdx.y * 128 + wm * 32;   // 2 M-subtiles of 16
    const int n0 = blockIdx.x * 128 + wn * 64;   // 4 N-subtiles of 16

    v8f acc[2][4] = {};

    for (int k0 = 0; k0 < K; k0 += 32) {
        v16h a[2], b[4];
        // A fragment 16x32: lanes 0-15 (M=ll): K = k0..k0+7 , k0+16..k0+23
        //                   lanes 16-31      : K = k0+8..15 , k0+24..31
        #pragma unroll
        for (int i = 0; i < 2; i++) {
            const _Float16* pa = A + (size_t)(m0 + i * 16 + ll) * lda + (k0 + hi * 8);
            v8h p0 = *(const v8h*)(pa);
            v8h p1 = *(const v8h*)(pa + 16);
            #pragma unroll
            for (int e = 0; e < 8; e++) { a[i][e] = p0[e]; a[i][8 + e] = p1[e]; }
        }
        // B fragment 32x16 from Bt[N][K]: lane column n=ll, K split like A (k0+hi*8.., k0+16+hi*8..)
        #pragma unroll
        for (int j = 0; j < 4; j++) {
            int n = n0 + j * 16 + ll; n = (n < N) ? n : (N - 1);   // clamp (dup, discarded at store)
            const _Float16* pb = Bt + (size_t)n * ldb + (k0 + hi * 8);
            v8h q0 = *(const v8h*)(pb);
            v8h q1 = *(const v8h*)(pb + 16);
            #pragma unroll
            for (int e = 0; e < 8; e++) { b[j][e] = q0[e]; b[j][8 + e] = q1[e]; }
        }
        #pragma unroll
        for (int i = 0; i < 2; i++)
            #pragma unroll
            for (int j = 0; j < 4; j++)
                acc[i][j] = __builtin_amdgcn_wmma_f32_16x16x32_f16(
                    false, a[i], false, b[j], (short)0, acc[i][j], false, false);
    }

    // Store + epilogue.  C/D layout: VGPR r -> M = m0+i*16+hi*8+r, N = n0+j*16+ll
    #pragma unroll
    for (int i = 0; i < 2; i++) {
        #pragma unroll
        for (int j = 0; j < 4; j++) {
            int n = n0 + j * 16 + ll;
            if (n >= N) continue;
            #pragma unroll
            for (int r = 0; r < 8; r++) {
                int m = m0 + i * 16 + hi * 8 + r;
                float v = acc[i][j][r];
                size_t idx = (size_t)m * ldc + n;
                if      (EP == 0) Cf[idx] = v;                                   // plain f32
                else if (EP == 1) Cf[idx] = 1.f / (1.f + __expf(-v));            // sigmoid
                else if (EP == 2) Cf[idx] = v / (1.f + __expf(-v));              // silu
                else if (EP == 3) Ch[idx] = (_Float16)tanhf(v);                  // tanh -> f16
                else if (EP == 4) { float t = v > 0.f ? v : 0.f;                 // relu^2 -> f16
                                    Ch[idx] = (_Float16)(t * t); }
                else if (EP == 5) Cf[idx] = __expf(-__expf(e0[n] + v));          // rwkv decay
                else if (EP == 6) Cf[idx] += v;                                  // residual add
                else if (EP == 7) Cf[idx] += e0[idx] * v;                        // x += r * acc
                else if (EP == 9) Ch[idx] = (_Float16)(e1[idx] + e2[idx] * (e0[n] + v)); // maa mix
            }
        }
    }
}

// ---------------------------------------------------------------------------
// Elementwise / normalization kernels
// ---------------------------------------------------------------------------
__global__ __launch_bounds__(256)
void ln_kernel(const float* __restrict__ x, const float* __restrict__ w,
               float* __restrict__ out)
{
    __shared__ float sa[256], sb[256];
    const int row = blockIdx.x;
    const float* xr = x + (size_t)row * Dc;
    float s = 0.f, q = 0.f;
    for (int c = threadIdx.x; c < Dc; c += 256) { float v = xr[c]; s += v; q += v * v; }
    sa[threadIdx.x] = s; sb[threadIdx.x] = q;
    __syncthreads();
    for (int off = 128; off > 0; off >>= 1) {
        if ((int)threadIdx.x < off) { sa[threadIdx.x] += sa[threadIdx.x + off];
                                      sb[threadIdx.x] += sb[threadIdx.x + off]; }
        __syncthreads();
    }
    float mu  = sa[0] * (1.f / Dc);
    float var = sb[0] * (1.f / Dc) - mu * mu;
    float rs  = rsqrtf(var + EPSc);
    for (int c = threadIdx.x; c < Dc; c += 256)
        out[(size_t)row * Dc + c] = (xr[c] - mu) * rs * w[c];
}

__global__ void copy_kernel(const float* __restrict__ src, float* __restrict__ dst, size_t tot)
{
    for (size_t i = (size_t)blockIdx.x * blockDim.x + threadIdx.x; i < tot;
         i += (size_t)gridDim.x * blockDim.x) dst[i] = src[i];
}

__global__ void shift_kernel(const float* __restrict__ xln, float* __restrict__ sx)
{
    const size_t tot = (size_t)BTc * Dc;
    for (size_t i = (size_t)blockIdx.x * blockDim.x + threadIdx.x; i < tot;
         i += (size_t)gridDim.x * blockDim.x) {
        size_t bt = i / Dc;
        int t = (int)(bt % Tc);
        float prev = (t > 0) ? xln[i - Dc] : 0.f;
        sx[i] = prev - xln[i];
    }
}

__global__ void mix16_kernel(const float* __restrict__ xln, const float* __restrict__ sx,
                             const float* __restrict__ maa, _Float16* __restrict__ out)
{
    const size_t tot = (size_t)BTc * Dc;
    for (size_t i = (size_t)blockIdx.x * blockDim.x + threadIdx.x; i < tot;
         i += (size_t)gridDim.x * blockDim.x)
        out[i] = (_Float16)(xln[i] + sx[i] * maa[i & (Dc - 1)]);
}

// f32 [K][N]  ->  f16 transposed [N][K]
__global__ void convT_kernel(const float* __restrict__ src, _Float16* __restrict__ dst,
                             int K, int N)
{
    const size_t tot = (size_t)K * N;
    for (size_t i = (size_t)blockIdx.x * blockDim.x + threadIdx.x; i < tot;
         i += (size_t)gridDim.x * blockDim.x) {
        size_t n = i / (size_t)K, k = i % (size_t)K;
        dst[i] = (_Float16)src[k * (size_t)N + n];
    }
}

// GroupNorm over each head's 64 channels + affine + gate multiply -> f16
__global__ __launch_bounds__(64)
void gn_gate_kernel(const float* __restrict__ wkv, const float* __restrict__ g,
                    const float* __restrict__ gnw, const float* __restrict__ gnb,
                    _Float16* __restrict__ att)
{
    __shared__ float sa[64], sb[64];
    int row = blockIdx.x >> 4;
    int h   = blockIdx.x & 15;
    int j   = threadIdx.x;
    int d   = h * HSc + j;
    size_t idx = (size_t)row * Dc + d;
    float v = wkv[idx];
    sa[j] = v; sb[j] = v * v;
    __syncthreads();
    for (int off = 32; off > 0; off >>= 1) {
        if (j < off) { sa[j] += sa[j + off]; sb[j] += sb[j + off]; }
        __syncthreads();
    }
    float mu  = sa[0] * (1.f / HSc);
    float var = sb[0] * (1.f / HSc) - mu * mu;
    float y   = (v - mu) * rsqrtf(var + EPSc) * gnw[d] + gnb[d];
    att[idx] = (_Float16)(y * g[idx]);
}

// ---------------------------------------------------------------------------
// Chunked WKV scan.  Layout of r/k/v/w: [B*T, D] with head h at cols h*64..
// state[k][v]; thread = v column, 64 state registers per thread.
// ---------------------------------------------------------------------------
__global__ __launch_bounds__(64)
void wkv_pass1(const float* __restrict__ k, const float* __restrict__ v,
               const float* __restrict__ w, float* __restrict__ S)
{
    int bid = blockIdx.x;
    int c  = bid % NCc;
    int bh = bid / NCc;
    int h  = bh & 15, b = bh >> 4;
    int vi = threadIdx.x;
    float s[HSc];
    #pragma unroll
    for (int kk = 0; kk < HSc; kk++) s[kk] = 0.f;
    __shared__ float lk[HSc], lw[HSc], lv[HSc];
    for (int tt = 0; tt < CTc; tt++) {
        int t = c * CTc + tt;
        size_t base = ((size_t)(b * Tc + t)) * Dc + h * HSc;
        lk[vi] = k[base + vi]; lw[vi] = w[base + vi]; lv[vi] = v[base + vi];
        __syncthreads();
        float vt = lv[vi];
        #pragma unroll
        for (int kk = 0; kk < HSc; kk++) s[kk] = s[kk] * lw[kk] + lk[kk] * vt;
        __syncthreads();
    }
    size_t so = (size_t)bid * HSc * HSc;
    #pragma unroll
    for (int kk = 0; kk < HSc; kk++) S[so + kk * HSc + vi] = s[kk];
}

__global__ __launch_bounds__(64)
void wkv_pprod(const float* __restrict__ w, float* __restrict__ P)
{
    int bh = blockIdx.x;
    int h = bh & 15, b = bh >> 4;
    int kk = threadIdx.x;
    for (int c = 0; c < NCc; c++) {
        float p = 1.f;
        for (int tt = 0; tt < CTc; tt++) {
            int t = c * CTc + tt;
            p *= w[((size_t)(b * Tc + t)) * Dc + h * HSc + kk];
        }
        P[((size_t)bh * NCc + c) * HSc + kk] = p;
    }
}

__global__ __launch_bounds__(64)
void wkv_pass2(const float* __restrict__ S, const float* __restrict__ P,
               float* __restrict__ Init)
{
    int bh = blockIdx.x;
    int vi = threadIdx.x;
    float st[HSc];
    #pragma unroll
    for (int kk = 0; kk < HSc; kk++) st[kk] = 0.f;
    __shared__ float lp[HSc];
    for (int c = 0; c < NCc; c++) {
        size_t so = ((size_t)(bh * NCc + c)) * HSc * HSc;
        #pragma unroll
        for (int kk = 0; kk < HSc; kk++) Init[so + kk * HSc + vi] = st[kk];
        lp[vi] = P[((size_t)bh * NCc + c) * HSc + vi];
        __syncthreads();
        #pragma unroll
        for (int kk = 0; kk < HSc; kk++) st[kk] = st[kk] * lp[kk] + S[so + kk * HSc + vi];
        __syncthreads();
    }
}

__global__ __launch_bounds__(64)
void wkv_pass3(const float* __restrict__ r, const float* __restrict__ k,
               const float* __restrict__ v, const float* __restrict__ w,
               const float* __restrict__ u, const float* __restrict__ Init,
               float* __restrict__ out)
{
    int bid = blockIdx.x;
    int c  = bid % NCc;
    int bh = bid / NCc;
    int h  = bh & 15, b = bh >> 4;
    int vi = threadIdx.x;
    float st[HSc];
    size_t so = (size_t)bid * HSc * HSc;
    #pragma unroll
    for (int kk = 0; kk < HSc; kk++) st[kk] = Init[so + kk * HSc + vi];
    __shared__ float lr[HSc], lk[HSc], lw[HSc], lv[HSc], lu[HSc];
    lu[vi] = u[h * HSc + vi];
    __syncthreads();
    for (int tt = 0; tt < CTc; tt++) {
        int t = c * CTc + tt;
        size_t base = ((size_t)(b * Tc + t)) * Dc + h * HSc;
        lr[vi] = r[base + vi]; lk[vi] = k[base + vi];
        lw[vi] = w[base + vi]; lv[vi] = v[base + vi];
        __syncthreads();
        float vt = lv[vi];
        float o = 0.f;
        #pragma unroll
        for (int kk = 0; kk < HSc; kk++) {
            float kv = lk[kk] * vt;
            o += lr[kk] * (st[kk] + lu[kk] * kv);
            st[kk] = st[kk] * lw[kk] + kv;
        }
        out[base + vi] = o;
        __syncthreads();
    }
}

// ---------------------------------------------------------------------------
// Host orchestration
// ---------------------------------------------------------------------------
#define GEMM(EP, A_, lda_, Bt_, ldb_, Cf_, Ch_, ldc_, N_, K_, e0_, e1_, e2_)          \
    gemm_wmma_f16<EP><<<dim3(((N_) + 127) / 128, BTc / 128), dim3(256), 0, stream>>>( \
        A_, lda_, Bt_, ldb_, Cf_, Ch_, ldc_, BTc, N_, K_, e0_, e1_, e2_)

extern "C" void kernel_launch(void* const* d_in, const int* in_sizes, int n_in,
                              void* d_out, int out_size, void* d_ws, size_t ws_size,
                              hipStream_t stream)
{
    (void)in_sizes; (void)n_in; (void)out_size; (void)ws_size;
    const float* Xin   = (const float*)d_in[0];
    const float* LN1   = (const float*)d_in[1];
    const float* LN2   = (const float*)d_in[2];
    const float* XMAA  = (const float*)d_in[3];
    const float* WMAA  = (const float*)d_in[4];
    const float* KMAA  = (const float*)d_in[5];
    const float* VMAA  = (const float*)d_in[6];
    const float* RMAA  = (const float*)d_in[7];
    const float* GMAA  = (const float*)d_in[8];
    const float* TMW1  = (const float*)d_in[9];
    const float* TMW2  = (const float*)d_in[10];
    const float* TDW1  = (const float*)d_in[11];
    const float* TDW2  = (const float*)d_in[12];
    const float* TDEC  = (const float*)d_in[13];
    const float* TFIR  = (const float*)d_in[14];
    const float* WR    = (const float*)d_in[15];
    const float* WK    = (const float*)d_in[16];
    const float* WV    = (const float*)d_in[17];
    const float* WO    = (const float*)d_in[18];
    const float* WG    = (const float*)d_in[19];
    const float* GNW   = (const float*)d_in[20];
    const float* GNB   = (const float*)d_in[21];
    const float* CKMAA = (const float*)d_in[22];
    const float* CRMAA = (const float*)d_in[23];
    const float* CWK   = (const float*)d_in[24];
    const float* CWV   = (const float*)d_in[25];
    const float* CWR   = (const float*)d_in[26];
    const float* LNO   = (const float*)d_in[27];

    char* ws = (char*)d_ws;
    size_t off = 0;
    auto take = [&](size_t bytes) -> char* {
        char* p = ws + off;
        off = (off + bytes + 255) & ~(size_t)255;
        return p;
    };
    const size_t NF = (size_t)BTc * Dc;   // 8M elements
    float* xst  = (float*)take(NF * 4);   // residual stream
    float* xln  = (float*)take(NF * 4);
    float* sx   = (float*)take(NF * 4);
    float* rb   = (float*)take(NF * 4);
    float* kb   = (float*)take(NF * 4);
    float* vb   = (float*)take(NF * 4);
    float* gb   = (float*)take(NF * 4);
    float* wb   = (float*)take(NF * 4);
    float* wkvo = (float*)take(NF * 4);
    _Float16* xxx16 = (_Float16*)take(NF * 2);
    _Float16* a16[5];
    for (int i = 0; i < 5; i++) a16[i] = (_Float16*)take(NF * 2);
    _Float16* mixb  = (_Float16*)take((size_t)BTc * 160 * 2);
    _Float16* wmid  = (_Float16*)take((size_t)BTc * 64 * 2);
    _Float16* att16 = (_Float16*)take(NF * 2);
    _Float16* kc16  = (_Float16*)take((size_t)BTc * 3072 * 2);
    float* Sb = (float*)take((size_t)Bc * Hc * NCc * HSc * HSc * 4);
    float* Ib = (float*)take((size_t)Bc * Hc * NCc * HSc * HSc * 4);
    float* Pb = (float*)take((size_t)Bc * Hc * NCc * HSc * 4);
    _Float16* WrT  = (_Float16*)take((size_t)Dc * Dc * 2);
    _Float16* WkT  = (_Float16*)take((size_t)Dc * Dc * 2);
    _Float16* WvT  = (_Float16*)take((size_t)Dc * Dc * 2);
    _Float16* WgT  = (_Float16*)take((size_t)Dc * Dc * 2);
    _Float16* WoT  = (_Float16*)take((size_t)Dc * Dc * 2);
    _Float16* tw1T = (_Float16*)take((size_t)160 * Dc * 2);
    _Float16* tw2T = (_Float16*)take((size_t)5 * Dc * Ec * 2);
    _Float16* td1T = (_Float16*)take((size_t)EDc * Dc * 2);
    _Float16* td2T = (_Float16*)take((size_t)Dc * EDc * 2);
    _Float16* cWkT = (_Float16*)take((size_t)3 * Dc * Dc * 2);
    _Float16* cWvT = (_Float16*)take((size_t)Dc * 3 * Dc * 2);
    _Float16* cWrT = (_Float16*)take((size_t)Dc * Dc * 2);

    const dim3 EB(256), EG(4096);

    copy_kernel<<<EG, EB, 0, stream>>>(Xin, xst, NF);

    for (int l = 0; l < Lc; l++) {
        const size_t oD  = (size_t)l * Dc;
        const size_t oDD = (size_t)l * Dc * Dc;

        // --- weight conversion: f32 [K][N] -> f16 transposed [N][K] ---
        convT_kernel<<<EG, EB, 0, stream>>>(WR + oDD, WrT, Dc, Dc);
        convT_kernel<<<EG, EB, 0, stream>>>(WK + oDD, WkT, Dc, Dc);
        convT_kernel<<<EG, EB, 0, stream>>>(WV + oDD, WvT, Dc, Dc);
        convT_kernel<<<EG, EB, 0, stream>>>(WG + oDD, WgT, Dc, Dc);
        convT_kernel<<<EG, EB, 0, stream>>>(WO + oDD, WoT, Dc, Dc);
        convT_kernel<<<2048, EB, 0, stream>>>(TMW1 + (size_t)l * Dc * 5 * Ec, tw1T, Dc, 5 * Ec);
        for (int f = 0; f < 5; f++)
            convT_kernel<<<256, EB, 0, stream>>>(TMW2 + ((size_t)l * 5 + f) * Ec * Dc,
                                                 tw2T + (size_t)f * Dc * Ec, Ec, Dc);
        convT_kernel<<<512, EB, 0, stream>>>(TDW1 + (size_t)l * Dc * EDc, td1T, Dc, EDc);
        convT_kernel<<<512, EB, 0, stream>>>(TDW2 + (size_t)l * EDc * Dc, td2T, EDc, Dc);
        convT_kernel<<<EG, EB, 0, stream>>>(CWK + (size_t)l * Dc * 3 * Dc, cWkT, Dc, 3 * Dc);
        convT_kernel<<<EG, EB, 0, stream>>>(CWV + (size_t)l * 3 * Dc * Dc, cWvT, 3 * Dc, Dc);
        convT_kernel<<<EG, EB, 0, stream>>>(CWR + oDD, cWrT, Dc, Dc);

        // --- time mix ---
        ln_kernel<<<BTc, EB, 0, stream>>>(xst, LN1 + oD, xln);
        shift_kernel<<<EG, EB, 0, stream>>>(xln, sx);
        mix16_kernel<<<EG, EB, 0, stream>>>(xln, sx, XMAA + oD, xxx16);
        // mix = tanh(xxx @ tm_w1)  [BT,160] f16
        GEMM(3, xxx16, Dc, tw1T, Dc, (float*)nullptr, mixb, 5 * Ec, 5 * Ec, Dc,
             (const float*)nullptr, (const float*)nullptr, (const float*)nullptr);
        // five data-dependent mixes -> f16 GEMM inputs (wx,kx,vx,rx,gx)
        const float* maas[5] = {WMAA + oD, KMAA + oD, VMAA + oD, RMAA + oD, GMAA + oD};
        for (int f = 0; f < 5; f++)
            GEMM(9, mixb + f * Ec, 5 * Ec, tw2T + (size_t)f * Dc * Ec, Ec,
                 (float*)nullptr, a16[f], Dc, Dc, Ec, maas[f], xln, sx);
        GEMM(1, a16[3], Dc, WrT, Dc, rb, (_Float16*)nullptr, Dc, Dc, Dc,
             (const float*)nullptr, (const float*)nullptr, (const float*)nullptr);  // r = sigmoid
        GEMM(0, a16[1], Dc, WkT, Dc, kb, (_Float16*)nullptr, Dc, Dc, Dc,
             (const float*)nullptr, (const float*)nullptr, (const float*)nullptr);  // k
        GEMM(0, a16[2], Dc, WvT, Dc, vb, (_Float16*)nullptr, Dc, Dc, Dc,
             (const float*)nullptr, (const float*)nullptr, (const float*)nullptr);  // v
        GEMM(2, a16[4], Dc, WgT, Dc, gb, (_Float16*)nullptr, Dc, Dc, Dc,
             (const float*)nullptr, (const float*)nullptr, (const float*)nullptr);  // g = silu
        GEMM(3, a16[0], Dc, td1T, Dc, (float*)nullptr, wmid, EDc, EDc, Dc,
             (const float*)nullptr, (const float*)nullptr, (const float*)nullptr);  // tanh(wx@td_w1)
        GEMM(5, wmid, EDc, td2T, EDc, wb, (_Float16*)nullptr, Dc, Dc, EDc,
             TDEC + (size_t)l * Hc * HSc, (const float*)nullptr, (const float*)nullptr); // decay

        // --- chunked wkv scan ---
        wkv_pass1<<<Bc * Hc * NCc, 64, 0, stream>>>(kb, vb, wb, Sb);
        wkv_pprod<<<Bc * Hc, 64, 0, stream>>>(wb, Pb);
        wkv_pass2<<<Bc * Hc, 64, 0, stream>>>(Sb, Pb, Ib);
        wkv_pass3<<<Bc * Hc * NCc, 64, 0, stream>>>(rb, kb, vb, wb,
                                                    TFIR + (size_t)l * Hc * HSc, Ib, wkvo);
        gn_gate_kernel<<<BTc * Hc, 64, 0, stream>>>(wkvo, gb, GNW + oD, GNB + oD, att16);
        // x += att @ Wo
        GEMM(6, att16, Dc, WoT, Dc, xst, (_Float16*)nullptr, Dc, Dc, Dc,
             (const float*)nullptr, (const float*)nullptr, (const float*)nullptr);

        // --- channel mix ---
        ln_kernel<<<BTc, EB, 0, stream>>>(xst, LN2 + oD, xln);
        shift_kernel<<<EG, EB, 0, stream>>>(xln, sx);
        mix16_kernel<<<EG, EB, 0, stream>>>(xln, sx, CKMAA + oD, a16[0]);
        mix16_kernel<<<EG, EB, 0, stream>>>(xln, sx, CRMAA + oD, a16[1]);
        // kc = relu(kx @ cWk)^2 -> f16
        GEMM(4, a16[0], Dc, cWkT, Dc, (float*)nullptr, kc16, 3 * Dc, 3 * Dc, Dc,
             (const float*)nullptr, (const float*)nullptr, (const float*)nullptr);
        // r = sigmoid(rx @ cWr)
        GEMM(1, a16[1], Dc, cWrT, Dc, rb, (_Float16*)nullptr, Dc, Dc, Dc,
             (const float*)nullptr, (const float*)nullptr, (const float*)nullptr);
        // x += r * (kc @ cWv)
        GEMM(7, kc16, 3 * Dc, cWvT, 3 * Dc, xst, (_Float16*)nullptr, Dc, Dc, 3 * Dc,
             rb, (const float*)nullptr, (const float*)nullptr);
    }

    ln_kernel<<<BTc, EB, 0, stream>>>(xst, LNO, (float*)d_out);
}
